// SiamFCXCorr_80942953660996
// MI455X (gfx1250) — matrix operand order repaired and checked by reading, not compiled, not run on hardware
//
#include <hip/hip_runtime.h>
#include <hip/hip_bf16.h>
#include <stdint.h>

// ---------------------------------------------------------------------------
// SiamFC: conv3x3+BN (z and x branches) via v_wmma_f32_16x16x32_bf16 implicit
// GEMM (double-buffered LDS im2col, scalar offset tables, uniform-base 32-bit
// vaddr gathers), then depthwise cross-correlation via LDS-staged f32 FMA.
// ---------------------------------------------------------------------------

typedef __attribute__((ext_vector_type(16))) __bf16 v16bf;
typedef __attribute__((ext_vector_type(8)))  float  v8f;

#define BATCH 64
#define CIN   256
#define CH    256
#define KTOT  2304            // CIN * 9
#define NKT   72              // KTOT / 32 (K chunks per WMMA step)
#define WPERSET (NKT * 16 * 32 * 8)   // swizzled-weight dwords per branch = 294912

__device__ __forceinline__ unsigned int pack_bf16(float lo, float hi) {
    // round-to-nearest-even f32 -> bf16, packed pair (lo in [15:0], hi in [31:16])
    unsigned int ul = __float_as_uint(lo);
    unsigned int uh = __float_as_uint(hi);
    ul = (ul + 0x7FFFu + ((ul >> 16) & 1u)) >> 16;
    uh = (uh + 0x7FFFu + ((uh >> 16) & 1u)) >> 16;
    return (ul & 0xFFFFu) | (uh << 16);
}

// ---------------------------------------------------------------------------
// Swizzle OIHW f32 weights into the exact WMMA bf16 B-fragment VGPR image:
//   dword index = ((ktile*16 + cotile)*32 + lane)*8 + v
//   lane L holds column n = L%16; K half = (L>=16)?16:0; VGPR v holds k = half+2v, +1
// so a wave loads a whole B frag as two coalesced b128 loads per lane.
// ---------------------------------------------------------------------------
__global__ void prep_weights(const float* __restrict__ w_z,
                             const float* __restrict__ w_x,
                             unsigned int* __restrict__ wt2) {
    unsigned int t = blockIdx.x * blockDim.x + threadIdx.x;
    if (t >= 2u * WPERSET) return;
    const float* w = (t < WPERSET) ? w_z : w_x;
    unsigned int* o = wt2 + ((t < WPERSET) ? 0u : (unsigned)WPERSET);
    unsigned int tt = (t < WPERSET) ? t : (t - WPERSET);

    unsigned int v    = tt & 7u;
    unsigned int lane = (tt >> 3) & 31u;
    unsigned int cot  = (tt >> 8) & 15u;
    unsigned int kt   = tt >> 12;

    unsigned int co = cot * 16u + (lane & 15u);
    unsigned int k0 = kt * 32u + ((lane >= 16u) ? 16u : 0u) + 2u * v;

    float f[2];
#pragma unroll
    for (int j = 0; j < 2; ++j) {
        unsigned int k   = k0 + (unsigned)j;      // k = ci*9 + r*3 + s
        unsigned int ci  = k / 9u;
        unsigned int rem = k - ci * 9u;
        unsigned int r   = rem / 3u;
        unsigned int s   = rem - r * 3u;
        f[j] = w[((co * CIN + ci) * 3u + r) * 3u + s];
    }
    o[tt] = pack_bf16(f[0], f[1]);
}

// Per-branch im2col offset tables: koffs[k] = (ci*H + r)*W + s
__global__ void prep_offs(int* __restrict__ koffs) {   // [2][KTOT]: z then x
    int t = blockIdx.x * blockDim.x + threadIdx.x;
    if (t >= 2 * KTOT) return;
    int set = (t < KTOT) ? 0 : 1;
    int k   = t - set * KTOT;
    int H   = set ? 32 : 16;
    int ci  = k / 9;
    int rem = k - ci * 9;
    int r   = rem / 3;
    int s   = rem - r * 3;
    koffs[t] = (ci * H + r) * H + s;   // W == H for both branches
}

// BN folding: out[0..255] = gamma*rsqrt(var+eps); out[256..511] = beta - mean*inv
__global__ void prep_bn(const float* __restrict__ g, const float* __restrict__ b,
                        const float* __restrict__ m, const float* __restrict__ v,
                        float* __restrict__ out) {
    int c = threadIdx.x;
    if (c < CH) {
        float inv = g[c] * rsqrtf(v[c] + 1e-5f);
        out[c]      = inv;
        out[CH + c] = b[c] - m[c] * inv;
    }
}

// im2col gather+pack: block-uniform base + scalar koffs -> 32-bit vaddr loads
__device__ __forceinline__ void stage_chunk(const float* __restrict__ inb,
                                            unsigned pixoff,
                                            const int* __restrict__ kof8, // scalar
                                            unsigned int* __restrict__ dst) {
    const int4* kp = (const int4*)kof8;       // 2x s_load_b128 (wave-uniform)
    int4 o0 = kp[0], o1 = kp[1];
    float f0 = inb[pixoff + (unsigned)o0.x];
    float f1 = inb[pixoff + (unsigned)o0.y];
    float f2 = inb[pixoff + (unsigned)o0.z];
    float f3 = inb[pixoff + (unsigned)o0.w];
    float f4 = inb[pixoff + (unsigned)o1.x];
    float f5 = inb[pixoff + (unsigned)o1.y];
    float f6 = inb[pixoff + (unsigned)o1.z];
    float f7 = inb[pixoff + (unsigned)o1.w];
    dst[0] = pack_bf16(f0, f1);
    dst[1] = pack_bf16(f2, f3);
    dst[2] = pack_bf16(f4, f5);
    dst[3] = pack_bf16(f6, f7);
}

// ---------------------------------------------------------------------------
// Implicit-GEMM conv3x3 + fused BN.
// Block = 128 threads (4 waves). Block tile: 32 output pixels x 256 channels.
// Wave w: all 32 pixels (2 M-subtiles) x 64 channels (4 N-tiles) -> 8 WMMA/step.
// K loop: 72 chunks of 32, double-buffered LDS im2col (one barrier per step),
// fragment loads first, staging of next chunk overlaps the 8-WMMA burst.
// ---------------------------------------------------------------------------
__global__ __launch_bounds__(128) void conv_bn_wmma(
    const float* __restrict__ in,          // [B, CIN, H, W] f32
    const unsigned int* __restrict__ wt2,  // swizzled bf16 weight frags
    const int* __restrict__ koffs,         // [KTOT] im2col offsets
    const float* __restrict__ bn,          // [512] inv, bias
    float* __restrict__ outp,              // [B, CH, PIX] f32
    int H, int W, int OW, int PIX, int npixtiles)
{
    __shared__ unsigned int lds_a[2][32 * 17];   // [buf][pix][kk/2], pitch 17

    const int b    = blockIdx.x / npixtiles;
    const int pt   = blockIdx.x - b * npixtiles;
    const int tid  = threadIdx.x;
    const int lane = tid & 31;
    // wave id == k-group id; force scalar so offset-table reads become SMEM
    const int wvu  = __builtin_amdgcn_readfirstlane(tid >> 5);

    // --- im2col role: thread owns pixel = lane and 8 consecutive k values
    int P  = pt * 32 + lane;
    int Pc = (P < PIX) ? P : (PIX - 1);        // clamp padded pixels (not stored)
    int oy = Pc / OW;
    int ox = Pc - oy * OW;
    const float* inb = in + (size_t)b * CIN * H * W;   // block-uniform base
    const unsigned pixoff = (unsigned)(oy * W + ox);   // 32-bit per-lane offset
    const int ldsrow = lane * 17 + wvu * 4;    // this thread's 4 LDS dwords

    // --- A-frag role: A layout used here (16x32, contiguous k as in B): lane row m = L%16;
    // VGPR v holds packed (k, k+1), k = 2v + (L>=16?16:0)
    const int mlo = lane & 15;
    const int kh2 = (lane >> 4) * 8;           // (k-half offset) in dwords

    v8f acc[2][4];
#pragma unroll
    for (int ms = 0; ms < 2; ++ms)
#pragma unroll
        for (int n = 0; n < 4; ++n)
#pragma unroll
            for (int i = 0; i < 8; ++i) acc[ms][n][i] = 0.0f;

    const uint4* bp = (const uint4*)wt2;

    // prologue: stage K-chunk 0 into buffer 0
    stage_chunk(inb, pixoff, koffs + wvu * 8, &lds_a[0][ldsrow]);
    __syncthreads();

    for (int kt = 0; kt < NKT; ++kt) {
        const int cur = kt & 1;

        if (kt + 1 < NKT)   // hint next weight tile toward L2 (global_prefetch)
            __builtin_prefetch(&wt2[(((kt + 1) * 16 + wvu * 4) * 32) * 8], 0, 1);

        // A fragments for both M-subtiles (each lowers to wide ds_loads)
        union { unsigned int u[8]; v16bf v; } af[2];
#pragma unroll
        for (int ms = 0; ms < 2; ++ms) {
            const unsigned int* row = &lds_a[cur][(ms * 16 + mlo) * 17];
#pragma unroll
            for (int vv = 0; vv < 8; ++vv) {
                int d = vv + kh2;
                af[ms].u[vv] = row[d];
            }
        }

        // B fragments: 4 channel tiles, 2 coalesced b128 each (scalar base)
        union { uint4 q[2]; v16bf v; } bfr[4];
#pragma unroll
        for (int n = 0; n < 4; ++n) {
            int u4 = ((kt * 16 + (wvu * 4 + n)) * 32 + lane) * 2;
            bfr[n].q[0] = bp[u4];
            bfr[n].q[1] = bp[u4 + 1];
        }

        // stage next K-chunk into the other buffer (overlaps the WMMA burst)
        if (kt + 1 < NKT)
            stage_chunk(inb, pixoff, koffs + (kt + 1) * 32 + wvu * 8,
                        &lds_a[cur ^ 1][ldsrow]);

        // 8 back-to-back WMMAs
#pragma unroll
        for (int ms = 0; ms < 2; ++ms)
#pragma unroll
            for (int n = 0; n < 4; ++n)
                acc[ms][n] = __builtin_amdgcn_wmma_f32_16x16x32_bf16(
                    false, af[ms].v, false, bfr[n].v, (short)0, acc[ms][n], false, false);

        __syncthreads();
    }

    // --- epilogue: C layout (f32 16x16): n = lane%16, m = i + 8*(lane>=16)
    const int mBase = (lane >> 4) * 8;
    const int nC    = lane & 15;
#pragma unroll
    for (int n = 0; n < 4; ++n) {
        int   co   = (wvu * 4 + n) * 16 + nC;
        float inv  = bn[co];
        float bias = bn[CH + co];
        float* op  = outp + ((size_t)b * CH + co) * PIX;
#pragma unroll
        for (int ms = 0; ms < 2; ++ms)
#pragma unroll
            for (int i = 0; i < 8; ++i) {
                int P2 = pt * 32 + ms * 16 + mBase + i;
                if (P2 < PIX) op[P2] = acc[ms][n][i] * inv + bias;
            }
    }
}

// ---------------------------------------------------------------------------
// Depthwise xcorr: one block per (b,c). zf 14x14 slides over xf 30x30 -> 17x17.
// Not a shared-K GEMM (both operands depend on the channel) and only ~1% of
// the FLOPs, so VALU f32 FMA with LDS-staged planes is the right path.
// ---------------------------------------------------------------------------
__global__ __launch_bounds__(256) void xcorr_kernel(const float* __restrict__ zf,
                                                    const float* __restrict__ xf,
                                                    float* __restrict__ out) {
    __shared__ float sz[196];
    __shared__ float sx[900];
    const int bc  = blockIdx.x;
    const int tid = threadIdx.x;
    const float* zp = zf + (size_t)bc * 196;
    const float* xp = xf + (size_t)bc * 900;
    if (tid < 196) sz[tid] = zp[tid];
    for (int i = tid; i < 900; i += 256) sx[i] = xp[i];
    __syncthreads();

    for (int p = tid; p < 289; p += 256) {
        int oy = p / 17;
        int ox = p - oy * 17;
        float s = 0.0f;
        for (int ry = 0; ry < 14; ++ry) {
            const float* xr = &sx[(oy + ry) * 30 + ox];
            const float* zr = &sz[ry * 14];
#pragma unroll
            for (int rx = 0; rx < 14; ++rx) s = fmaf(zr[rx], xr[rx], s);
        }
        out[(size_t)bc * 289 + p] = s;
    }
}

// ---------------------------------------------------------------------------
extern "C" void kernel_launch(void* const* d_in, const int* in_sizes, int n_in,
                              void* d_out, int out_size, void* d_ws, size_t ws_size,
                              hipStream_t stream) {
    (void)in_sizes; (void)n_in; (void)out_size; (void)ws_size;

    const float* z   = (const float*)d_in[0];
    const float* x   = (const float*)d_in[1];
    const float* w_z = (const float*)d_in[2];
    const float* w_x = (const float*)d_in[3];
    const float* gz  = (const float*)d_in[4];
    const float* bz  = (const float*)d_in[5];
    const float* mz  = (const float*)d_in[6];
    const float* vz  = (const float*)d_in[7];
    const float* gx  = (const float*)d_in[8];
    const float* bx  = (const float*)d_in[9];
    const float* mx  = (const float*)d_in[10];
    const float* vx  = (const float*)d_in[11];

    // workspace layout
    char* ws = (char*)d_ws;
    unsigned int* wt2 = (unsigned int*)ws;                    // 2 * 1.18 MB
    float* bn_z  = (float*)(ws + 2ull * WPERSET * 4ull);
    float* bn_x  = bn_z + 512;
    int*   koffs = (int*)(bn_x + 512);                        // [2][KTOT]
    float* zf    = (float*)(koffs + 2 * KTOT);                // 64*256*196 f32
    float* xf    = zf + (size_t)BATCH * CH * 196;             // 64*256*900 f32
    float* outp  = (float*)d_out;                             // 64*256*289 f32

    prep_weights<<<(2 * WPERSET + 255) / 256, 256, 0, stream>>>(w_z, w_x, wt2);
    prep_offs<<<(2 * KTOT + 255) / 256, 256, 0, stream>>>(koffs);
    prep_bn<<<1, 256, 0, stream>>>(gz, bz, mz, vz, bn_z);
    prep_bn<<<1, 256, 0, stream>>>(gx, bx, mx, vx, bn_x);

    // z branch: 16x16 -> 14x14 (196 pix, 7 tiles of 32)
    conv_bn_wmma<<<BATCH * 7, 128, 0, stream>>>(z, wt2, koffs, bn_z, zf,
                                                16, 16, 14, 196, 7);
    // x branch: 32x32 -> 30x30 (900 pix, 29 tiles of 32)
    conv_bn_wmma<<<BATCH * 29, 128, 0, stream>>>(x, wt2 + WPERSET, koffs + KTOT,
                                                 bn_x, xf, 32, 32, 30, 900, 29);

    xcorr_kernel<<<BATCH * CH, 256, 0, stream>>>(zf, xf, outp);
}
